// AttLayer_74663711474018
// MI455X (gfx1250) — matrix operand order repaired and checked by reading, not compiled, not run on hardware
//
#include <hip/hip_runtime.h>

// ---------------------------------------------------------------------------
// AttnBlock for MI455X (gfx1250, wave32, WMMA 16x16x32 f16->f32, TDM staging)
// B=32, C=512, H=W=32 (N=1024 tokens), GROUPS=32
// ---------------------------------------------------------------------------

typedef _Float16 half_t;
typedef __attribute__((ext_vector_type(16))) _Float16 v16h;
typedef __attribute__((ext_vector_type(8)))  _Float16 v8h;
typedef __attribute__((ext_vector_type(8)))  float    v8f;
typedef __attribute__((ext_vector_type(4)))  float    v4f;
typedef __attribute__((ext_vector_type(4)))  unsigned v4u;
typedef __attribute__((ext_vector_type(8)))  int      v8i;
typedef __attribute__((ext_vector_type(4)))  int      v4i;

#define BATCH 32
#define CH    512
#define NTOK  1024
#define CTILES 16      // 512 / 32 (K per wmma)
#define SCALE 0.044194173824159216f  // 512^-0.5

// A-fragment (16x32 f16, M x K) for lane `l`: row m = l&15, half kh = l>>4.
// Lane holds K = {kh*8 .. +7} and {16+kh*8 .. +7}  (two 16B runs).
__device__ __forceinline__ v16h a_frag(const half_t* p, int kh) {
    v8h lo = *(const v8h*)(p + kh * 8);
    v8h hi = *(const v8h*)(p + 16 + kh * 8);
    return __builtin_shufflevector(lo, hi, 0,1,2,3,4,5,6,7,8,9,10,11,12,13,14,15);
}

// B-fragment (32x16 f16, K x N) for lane `l`: col n = l&15, half kh = l>>4.
// Lane holds K = {kh*8 .. +7} and {16+kh*8 .. +7}  (two 16B runs).
__device__ __forceinline__ v16h b_frag(const half_t* p, int kh) {
    v8h lo = *(const v8h*)(p + kh * 8);
    v8h hi = *(const v8h*)(p + 16 + kh * 8);
    return __builtin_shufflevector(lo, hi, 0,1,2,3,4,5,6,7,8,9,10,11,12,13,14,15);
}

__device__ __forceinline__ v8f wmma_f16(v16h a, v16h b, v8f c) {
    return __builtin_amdgcn_wmma_f32_16x16x32_f16(false, a, false, b, (short)0, c,
                                                  false, false);
}

__device__ __forceinline__ v8f v8f_zero() {
    v8f z = {0.f, 0.f, 0.f, 0.f, 0.f, 0.f, 0.f, 0.f};
    return z;
}

// ---------------------------------------------------------------------------
// K0: convert the four 512x512 f32 weights to f16 (packed wq|wk|wv|wp)
// ---------------------------------------------------------------------------
__global__ __launch_bounds__(256) void k_w2h(const float* __restrict__ wq,
                                             const float* __restrict__ wk,
                                             const float* __restrict__ wv,
                                             const float* __restrict__ wp,
                                             half_t* __restrict__ out) {
    int idx = blockIdx.x * 256 + threadIdx.x;          // 4 * 262144 total
    int which = idx >> 18;
    int r = idx & 262143;
    const float* src = (which == 0) ? wq : (which == 1) ? wk : (which == 2) ? wv : wp;
    out[idx] = (half_t)src[r];
}

// ---------------------------------------------------------------------------
// K1: GroupNorm(32) + affine, write token-major f16 hn[b][n][c]
// ---------------------------------------------------------------------------
__global__ __launch_bounds__(256) void k_groupnorm(const float* __restrict__ x,
                                                   const float* __restrict__ gw,
                                                   const float* __restrict__ gb,
                                                   half_t* __restrict__ hn) {
    int b = blockIdx.x >> 5;
    int g = blockIdx.x & 31;
    const float* xg = x + ((size_t)b * CH + g * 16) * NTOK;   // 16384 floats
    float s = 0.f, ss = 0.f;
    for (int i = threadIdx.x; i < 16384; i += 256) {
        float v = xg[i];
        s += v; ss += v * v;
    }
    __shared__ float red[24];
    for (int off = 16; off > 0; off >>= 1) {
        s  += __shfl_down(s,  off, 32);
        ss += __shfl_down(ss, off, 32);
    }
    int wave = threadIdx.x >> 5;
    if ((threadIdx.x & 31) == 0) { red[wave] = s; red[wave + 8] = ss; }
    __syncthreads();
    if (threadIdx.x == 0) {
        float ts = 0.f, tss = 0.f;
        for (int i = 0; i < 8; ++i) { ts += red[i]; tss += red[i + 8]; }
        float mu  = ts * (1.0f / 16384.0f);
        float var = tss * (1.0f / 16384.0f) - mu * mu;
        red[16] = mu;
        red[17] = rsqrtf(var + 1e-5f);
    }
    __syncthreads();
    float mu = red[16], rs = red[17];
    for (int i = threadIdx.x; i < 16384; i += 256) {
        int ch = i >> 10;
        int n  = i & 1023;
        int c  = g * 16 + ch;
        float v = (xg[i] - mu) * rs * gw[c] + gb[c];
        hn[((size_t)b * NTOK + n) * CH + c] = (half_t)v;
    }
}

// ---------------------------------------------------------------------------
// K2: Q/K/V projections; each wave produces a 16x64 strip (4 tiles, A reuse x4)
// grid = (64 tt * 8 og, B, 3), block = 1 wave.
// q,k stored [b][n][c]; v stored transposed [b][c][n] for the P·V GEMM.
// ---------------------------------------------------------------------------
__global__ __launch_bounds__(32) void k_qkv(const half_t* __restrict__ hn,
                                            const half_t* __restrict__ w16,
                                            const float* __restrict__ bq,
                                            const float* __restrict__ bk,
                                            const float* __restrict__ bv,
                                            half_t* __restrict__ q16,
                                            half_t* __restrict__ k16,
                                            half_t* __restrict__ vT16) {
    int b  = blockIdx.y;
    int z  = blockIdx.z;                  // 0=q 1=k 2=v
    int tt = blockIdx.x >> 3;             // token tile (0..63)
    int og = blockIdx.x & 7;              // out-channel group (0..7), 4 tiles each
    int lane = threadIdx.x;
    int ln = lane & 15, kh = lane >> 4;

    const half_t* A0 = hn + ((size_t)b * NTOK + tt * 16 + ln) * CH;
    const half_t* Wb = w16 + (size_t)z * (CH * CH) + (size_t)(og * 64 + ln) * CH;

    v8f acc[4];
    #pragma unroll
    for (int i = 0; i < 4; ++i) acc[i] = v8f_zero();

    for (int ct = 0; ct < CTILES; ++ct) {
        v16h a = a_frag(A0 + ct * 32, kh);               // reused x4
        #pragma unroll
        for (int i = 0; i < 4; ++i) {
            v16h bb = b_frag(Wb + (size_t)i * 16 * CH + ct * 32, kh);
            acc[i] = wmma_f16(a, bb, acc[i]);
        }
    }

    const float* bias = (z == 0) ? bq : (z == 1) ? bk : bv;
    #pragma unroll
    for (int i = 0; i < 4; ++i) {
        int o = og * 64 + i * 16 + ln;                   // D: N = ln
        float bo = bias[o];
        if (z < 2) {
            half_t* dst = (z == 0) ? q16 : k16;
            #pragma unroll
            for (int v = 0; v < 8; ++v) {
                int tok = tt * 16 + kh * 8 + v;          // D: M = v + kh*8
                dst[((size_t)b * NTOK + tok) * CH + o] = (half_t)(acc[i][v] + bo);
            }
        } else {
            half_t* dst = vT16 + ((size_t)b * CH + o) * NTOK + tt * 16 + kh * 8;
            #pragma unroll
            for (int v = 0; v < 8; ++v) dst[v] = (half_t)(acc[i][v] + bo);
        }
    }
}

// ---------------------------------------------------------------------------
// K3: fused attention for one (batch, 16-query tile).
// Stage the contiguous 16x512 Q strip (16 KB) into LDS with the Tensor Data
// Mover (one tensor_load_to_lds per block), then:
//   Phase 1: 8 waves x 8 k-tiles -> 16x1024 score strip in LDS (f32)
//   Phase 2: softmax (shuffle reductions + v_exp_f32), P -> f16 LDS
//   Phase 3: O = P @ V from transposed vT
// ---------------------------------------------------------------------------
__global__ __launch_bounds__(256) void k_attn(const half_t* __restrict__ q16,
                                              const half_t* __restrict__ k16,
                                              const half_t* __restrict__ vT16,
                                              half_t* __restrict__ O16) {
    __shared__ half_t Qs[16 * CH];       // 16 KB staged Q strip
    __shared__ float  Sf[16 * NTOK];     // 64 KB scores
    __shared__ half_t Ph[16 * NTOK];     // 32 KB probabilities
    int b  = blockIdx.y;
    int qt = blockIdx.x;
    int tid = threadIdx.x;
    int wave = tid >> 5;
    int lane = tid & 31;
    int ln = lane & 15, kh = lane >> 4;

    const half_t* qsrc = q16 + ((size_t)b * NTOK + qt * 16) * CH;  // 16 KB contiguous

#if defined(__has_builtin) && __has_builtin(__builtin_amdgcn_tensor_load_to_lds)
    if (wave == 0) {
        // Tensor DMA descriptor (ISA §8.3/§8.4): 1 x 8192 elem tile, 2B elems.
        unsigned lds_off = (unsigned)(size_t)(void*)&Qs[0];
        unsigned long long ga = (unsigned long long)(size_t)(const void*)qsrc;
        v4u g0;
        g0[0] = 1u;                                   // count=1, user mode
        g0[1] = lds_off;                              // lds_addr
        g0[2] = (unsigned)(ga & 0xFFFFFFFFull);       // global_addr[31:0]
        g0[3] = (unsigned)((ga >> 32) & 0x01FFFFFFull) | (2u << 30); // addr hi | type=2
        v8i g1;
        g1[0] = (int)(1u << 16);                      // data_size=1 (2 bytes), mask=0
        g1[1] = (int)(8192u << 16);                   // tensor_dim0[15:0] = 8192
        g1[2] = 0;                                    // tensor_dim0 hi, tensor_dim1 lo
        g1[3] = (int)(8192u << 16);                   // tile_dim0 = 8192
        g1[4] = 0;                                    // tile_dim1/2 unused
        g1[5] = 8192;                                 // tensor_dim0_stride
        g1[6] = 0;
        g1[7] = 0;
        v4i z4; z4[0] = 0; z4[1] = 0; z4[2] = 0; z4[3] = 0;
#if __clang_major__ >= 23
        v8i z8; for (int i = 0; i < 8; ++i) z8[i] = 0;
        __builtin_amdgcn_tensor_load_to_lds(g0, g1, z4, z4, z8, 0);
#else
        __builtin_amdgcn_tensor_load_to_lds(g0, g1, z4, z4, 0);
#endif
        __builtin_amdgcn_s_wait_tensorcnt(0);
    }
#else
    // Fallback: cooperative copy (16B per thread x 4 iters)
    for (int i = tid; i < 16 * CH / 8; i += 256)
        *((v8h*)Qs + i) = *((const v8h*)qsrc + i);
#endif
    __syncthreads();

    // ---- Phase 1: S = Q K^T * scale -------------------------------------
    const half_t* Qrow = &Qs[ln * CH];                   // A rows from LDS
    v8f acc[8];
    #pragma unroll
    for (int j = 0; j < 8; ++j) acc[j] = v8f_zero();

    for (int ct = 0; ct < CTILES; ++ct) {
        v16h a = a_frag(Qrow + ct * 32, kh);             // reused across 8 k-tiles
        #pragma unroll
        for (int j = 0; j < 8; ++j) {
            int kt = wave * 8 + j;
            const half_t* Krow = k16 + ((size_t)b * NTOK + kt * 16 + ln) * CH;
            v16h bb = b_frag(Krow + ct * 32, kh);
            acc[j] = wmma_f16(a, bb, acc[j]);
        }
    }
    #pragma unroll
    for (int j = 0; j < 8; ++j) {
        int kt = wave * 8 + j;
        #pragma unroll
        for (int v = 0; v < 8; ++v) {
            int m = v + kh * 8;
            Sf[m * NTOK + kt * 16 + ln] = acc[j][v] * SCALE;
        }
    }
    __syncthreads();

    // ---- Phase 2: softmax over each of the 16 rows ----------------------
    {
        int r = tid >> 4;        // row 0..15 (2 rows per wave)
        int j = tid & 15;        // column lane within row group
        float mx = -1e30f;
        for (int i = 0; i < 64; ++i)
            mx = fmaxf(mx, Sf[r * NTOK + j + i * 16]);
        for (int off = 8; off > 0; off >>= 1)
            mx = fmaxf(mx, __shfl_xor(mx, off, 32));
        float sum = 0.f;
        for (int i = 0; i < 64; ++i) {
            float e = __expf(Sf[r * NTOK + j + i * 16] - mx);
            Sf[r * NTOK + j + i * 16] = e;               // thread-private slots
            sum += e;
        }
        for (int off = 8; off > 0; off >>= 1)
            sum += __shfl_xor(sum, off, 32);
        float inv = 1.0f / sum;
        for (int i = 0; i < 64; ++i)
            Ph[r * NTOK + j + i * 16] = (half_t)(Sf[r * NTOK + j + i * 16] * inv);
    }
    __syncthreads();

    // ---- Phase 3: O = P @ V (wave owns 4 channel tiles) -----------------
    v8f acc2[4];
    #pragma unroll
    for (int i = 0; i < 4; ++i) acc2[i] = v8f_zero();

    for (int ks = 0; ks < 32; ++ks) {                    // K = 1024 tokens / 32
        v16h a = a_frag(&Ph[ln * NTOK + ks * 32], kh);   // from LDS, reused x4
        #pragma unroll
        for (int i = 0; i < 4; ++i) {
            int ct = wave * 4 + i;
            const half_t* Vrow =
                vT16 + ((size_t)b * CH + ct * 16 + ln) * NTOK + ks * 32;
            v16h bb = b_frag(Vrow, kh);
            acc2[i] = wmma_f16(a, bb, acc2[i]);
        }
    }
    #pragma unroll
    for (int i = 0; i < 4; ++i) {
        int ct = wave * 4 + i;
        #pragma unroll
        for (int v = 0; v < 8; ++v) {
            int m = v + kh * 8;
            O16[((size_t)b * NTOK + qt * 16 + m) * CH + ct * 16 + ln] = (half_t)acc2[i][v];
        }
    }
}

// ---------------------------------------------------------------------------
// K4: out = x + O @ wp^T + bp; each wave does 4 output tiles (A reuse x4).
// Written f32 in [B,C,H,W] (token contiguous -> float4 stores).
// ---------------------------------------------------------------------------
__global__ __launch_bounds__(32) void k_proj(const half_t* __restrict__ O16,
                                             const half_t* __restrict__ wp16,
                                             const float* __restrict__ bp,
                                             const float* __restrict__ x,
                                             float* __restrict__ out) {
    int b  = blockIdx.y;
    int tt = blockIdx.x >> 3;
    int og = blockIdx.x & 7;
    int lane = threadIdx.x;
    int ln = lane & 15, kh = lane >> 4;

    const half_t* A0 = O16 + ((size_t)b * NTOK + tt * 16 + ln) * CH;
    const half_t* Wb = wp16 + (size_t)(og * 64 + ln) * CH;

    v8f acc[4];
    #pragma unroll
    for (int i = 0; i < 4; ++i) acc[i] = v8f_zero();

    for (int ct = 0; ct < CTILES; ++ct) {
        v16h a = a_frag(A0 + ct * 32, kh);               // reused x4
        #pragma unroll
        for (int i = 0; i < 4; ++i) {
            v16h bb = b_frag(Wb + (size_t)i * 16 * CH + ct * 32, kh);
            acc[i] = wmma_f16(a, bb, acc[i]);
        }
    }

    #pragma unroll
    for (int i = 0; i < 4; ++i) {
        int o = og * 64 + i * 16 + ln;
        float bo = bp[o];
        size_t base = ((size_t)b * CH + o) * NTOK + tt * 16 + kh * 8;
        v4f x0 = *(const v4f*)(x + base);
        v4f x1 = *(const v4f*)(x + base + 4);
        v4f r0, r1;
        #pragma unroll
        for (int v = 0; v < 4; ++v) {
            r0[v] = x0[v] + acc[i][v] + bo;
            r1[v] = x1[v] + acc[i][v + 4] + bo;
        }
        *(v4f*)(out + base) = r0;
        *(v4f*)(out + base + 4) = r1;
    }
}

// ---------------------------------------------------------------------------
// Host launcher
// ---------------------------------------------------------------------------
extern "C" void kernel_launch(void* const* d_in, const int* in_sizes, int n_in,
                              void* d_out, int out_size, void* d_ws, size_t ws_size,
                              hipStream_t stream) {
    const float* x    = (const float*)d_in[0];
    const float* gn_w = (const float*)d_in[1];
    const float* gn_b = (const float*)d_in[2];
    const float* wq   = (const float*)d_in[3];
    const float* bq   = (const float*)d_in[4];
    const float* wk   = (const float*)d_in[5];
    const float* bk   = (const float*)d_in[6];
    const float* wv   = (const float*)d_in[7];
    const float* bv   = (const float*)d_in[8];
    const float* wp   = (const float*)d_in[9];
    const float* bp   = (const float*)d_in[10];
    float* out = (float*)d_out;

    // Workspace layout (f16 elements): 5 activation buffers of B*N*C + 4 weights
    const size_t BIG = (size_t)BATCH * NTOK * CH;        // 16,777,216 halves (32 MB)
    half_t* ws  = (half_t*)d_ws;
    half_t* hn  = ws;
    half_t* q16 = ws + 1 * BIG;
    half_t* k16 = ws + 2 * BIG;
    half_t* vT  = ws + 3 * BIG;
    half_t* O16 = ws + 4 * BIG;
    half_t* w16 = ws + 5 * BIG;                          // wq|wk|wv|wp f16

    k_w2h<<<4096, 256, 0, stream>>>(wq, wk, wv, wp, w16);
    k_groupnorm<<<BATCH * 32, 256, 0, stream>>>(x, gn_w, gn_b, hn);
    k_qkv<<<dim3(64 * 8, BATCH, 3), 32, 0, stream>>>(hn, w16, bq, bk, bv, q16, k16, vT);
    k_attn<<<dim3(64, BATCH), 256, 0, stream>>>(q16, k16, vT, O16);
    k_proj<<<dim3(64 * 8, BATCH), 32, 0, stream>>>(O16, w16 + 3 * (size_t)(CH * CH),
                                                   bp, x, out);
}
